// MyGraphConvolution_35794257445170
// MI455X (gfx1250) — compile-verified, matrix operand order checked by reading, not run on hardware
//
#include <hip/hip_runtime.h>
#include <hip/hip_bf16.h>

typedef __attribute__((ext_vector_type(16))) _Float16 v16h;
typedef __attribute__((ext_vector_type(8)))  _Float16 v8h;
typedef __attribute__((ext_vector_type(8)))  float    v8f;

#define NN    8192   // nodes
#define DIN   512    // input features
#define DOUT  256    // output features

#define ROWS  64     // output rows per block (stage 2)
#define KB    64     // K panel width (2 WMMA k-steps)
#define KPAD  72     // padded LDS row stride in f16 (144 B: 16B-aligned, bank-spread)

static __device__ inline v8f wmma_f16(const v16h& a, const v16h& b, const v8f& c) {
  return __builtin_amdgcn_wmma_f32_16x16x32_f16(false, a, false, b, (short)0, c,
                                                false, false);
}

// ---------------------------------------------------------------------------
// Stage 1: XWt[n][i] = sum_k x[i][k] * W[k][n]   (f16 output, transposed)
// ---------------------------------------------------------------------------
__global__ __launch_bounds__(256) void gcn_xw_t_f16(
    const float* __restrict__ x, const float* __restrict__ W,
    _Float16* __restrict__ xwt)
{
  const int lane = threadIdx.x & 31;
  const int wave = threadIdx.x >> 5;
  const int m    = lane & 15;
  const int half = lane >> 4;

  const int tile = blockIdx.x * 8 + wave;   // 8192 tiles total
  const int n0   = (tile & 15) * 16;        // DOUT block (rows of XWt)
  const int i0   = (tile >> 4) * 16;        // N block    (cols of XWt)

  v8f acc = {};
  for (int k0 = 0; k0 < DIN; k0 += 32) {
    v16h a;
#pragma unroll
    for (int e = 0; e < 8; ++e) {
      const int kA = k0 + half * 8 + e;
      a[e]     = (_Float16)W[kA * DOUT + n0 + m];
      a[e + 8] = (_Float16)W[(kA + 16) * DOUT + n0 + m];
    }
    const float* xp = x + (size_t)(i0 + m) * DIN + k0 + half * 8;
    float tb[16];
    *(float4*)(tb + 0)  = *(const float4*)(xp + 0);
    *(float4*)(tb + 4)  = *(const float4*)(xp + 4);
    *(float4*)(tb + 8)  = *(const float4*)(xp + 16);
    *(float4*)(tb + 12) = *(const float4*)(xp + 20);
    v16h b;
#pragma unroll
    for (int e = 0; e < 8; ++e) {
      b[e]     = (_Float16)tb[e];
      b[e + 8] = (_Float16)tb[8 + e];
    }
    acc = wmma_f16(a, b, acc);
  }
#pragma unroll
  for (int r = 0; r < 8; ++r) {
    const int nrow = n0 + half * 8 + r;
    xwt[(size_t)nrow * NN + i0 + m] = (_Float16)acc[r];
  }
}

// ---------------------------------------------------------------------------
// Stage 2: out = D^{-1} * A * XW.
// 512 threads (16 waves). Block owns 64 rows x 256 cols.
// Wave (rg, cg): rg = wave>>3 selects 32-row group, cg = wave&7 selects 32 cols.
// A panel (64 x 64 K) converted int32->f16 ONCE per block into LDS (dbl-buffered).
// Degree accumulated as ints during the fill phase.
// ---------------------------------------------------------------------------
__global__ __launch_bounds__(512) void gcn_agg_wmma(
    const int* __restrict__ adj, const _Float16* __restrict__ xwt,
    float* __restrict__ out)
{
  __shared__ __align__(16) _Float16 Apan[2][ROWS][KPAD];
  __shared__ int degb[ROWS];

  const int t    = threadIdx.x;
  const int lane = t & 31;
  const int wave = t >> 5;          // 0..15
  const int m    = lane & 15;
  const int half = lane >> 4;

  const int cg    = wave & 7;       // column group
  const int rg    = wave >> 3;      // row group
  const int n0    = cg * 32;        // wave's 32 output columns
  const int rbase = rg * 32;        // wave's 32 rows within block

  const int blockRow0 = blockIdx.x * ROWS;

  // fill-phase mapping: thread -> (row, 8-col slice) of the A panel
  const int frow = t >> 3;          // 0..63
  const int fcol = (t & 7) * 8;     // 0,8,...,56
  const int* fptr = adj + (size_t)(blockRow0 + frow) * NN + fcol;

  // B column pointers (xwt is [DOUT][NN], f16)
  const _Float16* bc0 = xwt + (size_t)(n0 + m) * NN;
  const _Float16* bc1 = xwt + (size_t)(n0 + 16 + m) * NN;

  v8f acc00 = {}, acc01 = {}, acc10 = {}, acc11 = {};
  int degp = 0;

  auto fill = [&](int buf, int k0) {
    int4 q0 = *(const int4*)(fptr + k0);
    int4 q1 = *(const int4*)(fptr + k0 + 4);
    __builtin_prefetch((const void*)(fptr + k0 + 2 * KB), 0, 0);
    unsigned v[8] = {(unsigned)q0.x, (unsigned)q0.y, (unsigned)q0.z, (unsigned)q0.w,
                     (unsigned)q1.x, (unsigned)q1.y, (unsigned)q1.z, (unsigned)q1.w};
    unsigned pk[4];
#pragma unroll
    for (int j = 0; j < 4; ++j) {
      const unsigned a = (unsigned)(v[2 * j] != 0u);
      const unsigned b = (unsigned)(v[2 * j + 1] != 0u);
      degp += (int)(a + b);
      pk[j] = (a | (b << 16)) * 0x3C00u;   // two f16 {0,1} packed
    }
    uint4 st = {pk[0], pk[1], pk[2], pk[3]};
    *(uint4*)&Apan[buf][frow][fcol] = st;  // ds_store_b128
  };

  auto loadA = [&](int buf, int R, int kk) {
    const _Float16* p = &Apan[buf][R + m][kk + half * 8];
    v8h lo = *(const v8h*)p;
    v8h hi = *(const v8h*)(p + 16);
    v16h a;
#pragma unroll
    for (int e = 0; e < 8; ++e) { a[e] = lo[e]; a[e + 8] = hi[e]; }
    return a;
  };

  auto loadB = [&](const _Float16* bc, int k) {
    const _Float16* p = bc + k + half * 8;
    v8h lo = *(const v8h*)p;
    v8h hi = *(const v8h*)(p + 16);
    v16h b;
#pragma unroll
    for (int e = 0; e < 8; ++e) { b[e] = lo[e]; b[e + 8] = hi[e]; }
    return b;
  };

  fill(0, 0);
  __syncthreads();

  constexpr int NITER = NN / KB;   // 128
  for (int it = 0; it < NITER; ++it) {
    const int k0  = it * KB;
    const int buf = it & 1;
    if (it + 1 < NITER) fill(buf ^ 1, k0 + KB);   // stage next panel
#pragma unroll
    for (int kk = 0; kk < KB; kk += 32) {
      v16h a0 = loadA(buf, rbase,      kk);
      v16h a1 = loadA(buf, rbase + 16, kk);
      v16h b0 = loadB(bc0, k0 + kk);
      v16h b1 = loadB(bc1, k0 + kk);
      acc00 = wmma_f16(a0, b0, acc00);
      acc01 = wmma_f16(a0, b1, acc01);
      acc10 = wmma_f16(a1, b0, acc10);
      acc11 = wmma_f16(a1, b1, acc11);
    }
    __syncthreads();
  }

  // Reduce degree over the 8 threads sharing a panel row (lane bits 0..2)
  degp += __shfl_xor(degp, 1, 32);
  degp += __shfl_xor(degp, 2, 32);
  degp += __shfl_xor(degp, 4, 32);
  if ((t & 7) == 0) degb[frow] = degp;
  __syncthreads();

  // C layout: element r holds M = half*8 + r, N = lane&15
#pragma unroll
  for (int r = 0; r < 8; ++r) {
    const int lr0 = rbase + half * 8 + r;        // tile-0 row in block
    const int lr1 = lr0 + 16;                    // tile-1 row in block
    const float inv0 = 1.0f / (float)degb[lr0];
    const float inv1 = 1.0f / (float)degb[lr1];
    const size_t gr0 = (size_t)(blockRow0 + lr0) * DOUT;
    const size_t gr1 = (size_t)(blockRow0 + lr1) * DOUT;
    out[gr0 + n0 + m]      = acc00[r] * inv0;
    out[gr0 + n0 + 16 + m] = acc01[r] * inv0;
    out[gr1 + n0 + m]      = acc10[r] * inv1;
    out[gr1 + n0 + 16 + m] = acc11[r] * inv1;
  }
}

// ---------------------------------------------------------------------------
extern "C" void kernel_launch(void* const* d_in, const int* in_sizes, int n_in,
                              void* d_out, int out_size, void* d_ws, size_t ws_size,
                              hipStream_t stream) {
  const float* x   = (const float*)d_in[0];   // [8192, 512] f32
  const int*   adj = (const int*)d_in[1];     // [8192, 8192] i32 (binary)
  const float* W   = (const float*)d_in[2];   // [512, 256] f32
  float*       out = (float*)d_out;           // [8192, 256] f32
  _Float16*    xwt = (_Float16*)d_ws;         // [256, 8192] f16 scratch (4 MB)

  gcn_xw_t_f16<<<dim3((NN / 16) * (DOUT / 16) / 8), 256, 0, stream>>>(x, W, xwt);
  gcn_agg_wmma<<<dim3(NN / ROWS), 512, 0, stream>>>(adj, xwt, out);
}
